// SchNet_GNN_Interact_41618233099045
// MI455X (gfx1250) — matrix operand order repaired and checked by reading, not compiled, not run on hardware
//
#include <hip/hip_runtime.h>
#include <hip/hip_bf16.h>

#define NNODES   40000
#define NGRAPHS  2000
#define ESCH     600000
#define ECHEM    80000
#define HIDD     128
#define NGAUSS   50
#define NODEDIM  39
#define BONDDIM  10
#define FPSIZE   2048
#define NTASKS   12
#define NLAYERS  6
#define CUTOFF_R 10.0f
#define PI_F     3.14159265358979323846f

typedef __attribute__((ext_vector_type(16))) _Float16 v16h;
typedef __attribute__((ext_vector_type(8)))  _Float16 v8h;
typedef __attribute__((ext_vector_type(8)))  float    v8f;
typedef __attribute__((ext_vector_type(4)))  float    v4f;

// ---------------------------------------------------------------- helpers

__device__ __forceinline__ float sspf(float x) {
    // shifted softplus, overflow-stable
    return fmaxf(x, 0.f) + log1pf(__expf(-fabsf(x))) - 0.6931471805599453f;
}

__device__ __forceinline__ v8h ld8(const _Float16* p) { return *(const v8h*)p; }

__device__ __forceinline__ v8h ld8(const float* p) {
    v4f a = *(const v4f*)p;
    v4f b = *(const v4f*)(p + 4);
    v8h r;
    r[0] = (_Float16)a[0]; r[1] = (_Float16)a[1]; r[2] = (_Float16)a[2]; r[3] = (_Float16)a[3];
    r[4] = (_Float16)b[0]; r[5] = (_Float16)b[1]; r[6] = (_Float16)b[2]; r[7] = (_Float16)b[3];
    return r;
}

// A fragment: 16x32 tile of A[M,K] (row-major, ld = K).
// lane l (m = l&15, g = l>>4) holds K = 16g + {0..15}.
template <typename TA>
__device__ __forceinline__ v16h afrag(const TA* __restrict__ A, const TA* __restrict__ A2,
                                      int ld, long m, int kb) {
    const TA* p = A + m * (long)ld + kb;
    v8h lo = ld8(p);
    v8h hi = ld8(p + 8);
    if (A2) {
        const TA* q = A2 + m * (long)ld + kb;
        v8h l2 = ld8(q), h2 = ld8(q + 8);
#pragma unroll
        for (int i = 0; i < 8; ++i) {
            lo[i] = (_Float16)((float)lo[i] + (float)l2[i]);
            hi[i] = (_Float16)((float)hi[i] + (float)h2[i]);
        }
    }
    v16h a;
#pragma unroll
    for (int i = 0; i < 8; ++i) { a[i] = lo[i]; a[8 + i] = hi[i]; }
    return a;
}

// B fragment: 32x16 tile taken from transposed weights Wt[N,Kpad] (row-major).
// lane l (n = l&15, g = l>>4) holds K = 16g + {0..15} -> one contiguous 32B run.
__device__ __forceinline__ v16h bfrag(const _Float16* __restrict__ Wt, int ld, long n, int kb) {
    return *(const v16h*)(Wt + n * (long)ld + kb);
}

// ---------------------------------------------------------------- GEMM (WMMA)
// D[M, NT*16] = act( (A (+A2)) @ Wt^T + bias (+res) ).  K % 32 == 0, M % 16 == 0.
// One wave computes a full 16 x (NT*16) output strip: A fragment loaded once per
// k-step and reused across NT WMMAs (8x A-traffic reduction at NT=8).
template <typename TA, typename TO, int NT>
__global__ void gemm_kernel(const TA* __restrict__ A, const TA* __restrict__ A2,
                            const _Float16* __restrict__ Wt, const float* __restrict__ bias,
                            const float* __restrict__ res, TO* __restrict__ D,
                            long strips, int K, int act) {
    const int N = NT * 16;
    long strip = (long)blockIdx.x * 8 + (threadIdx.x >> 5);
    if (strip >= strips) return;              // wave-uniform: EXEC stays full for WMMA
    int lane = threadIdx.x & 31;
    int grp = lane >> 4, lid = lane & 15;
    long mrow = strip * 16 + lid;

    const v8f zero = {};
    v8f acc[NT];
#pragma unroll
    for (int t = 0; t < NT; ++t) acc[t] = zero;

    for (int k0 = 0; k0 < K; k0 += 32) {
        v16h a = afrag<TA>(A, A2, K, mrow, k0 + 16 * grp);
#pragma unroll
        for (int t = 0; t < NT; ++t) {
            v16h b = bfrag(Wt, K, t * 16 + lid, k0 + 16 * grp);
            acc[t] = __builtin_amdgcn_wmma_f32_16x16x32_f16(false, a, false, b, (short)0,
                                                            acc[t], false, false);
        }
    }
#pragma unroll
    for (int t = 0; t < NT; ++t) {
        int ncol = t * 16 + lid;
        float bn = bias ? bias[ncol] : 0.f;
#pragma unroll
        for (int r = 0; r < 8; ++r) {
            long m = strip * 16 + r + grp * 8;
            float x = acc[t][r] + bn;
            if (res) x += res[m * (long)N + ncol];
            if (act == 1)      x = fmaxf(x, 0.f);
            else if (act == 2) x = sspf(x);
            D[m * (long)N + ncol] = (TO)x;
        }
    }
}

// Filter GEMM #2 (16x128 strip) with fused cutoff * gather * scatter-add epilogue:
// agg[col[e],:] += (ssp_hidden[e,:] @ W2 + b2) * ccos[e] * hv[row[e],:]
__global__ void gemm_filter_scatter_kernel(const _Float16* __restrict__ A,
                                           const _Float16* __restrict__ Wt,
                                           const float* __restrict__ bias,
                                           const float* __restrict__ ccos,
                                           const float* __restrict__ hv,
                                           const int* __restrict__ rowi,
                                           const int* __restrict__ coli,
                                           float* __restrict__ agg, long strips) {
    const int K = HIDD, NT = HIDD / 16;
    long strip = (long)blockIdx.x * 8 + (threadIdx.x >> 5);
    if (strip >= strips) return;
    int lane = threadIdx.x & 31;
    int grp = lane >> 4, lid = lane & 15;
    long mrow = strip * 16 + lid;

    const v8f zero = {};
    v8f acc[NT];
#pragma unroll
    for (int t = 0; t < NT; ++t) acc[t] = zero;

#pragma unroll
    for (int k0 = 0; k0 < K; k0 += 32) {
        v16h a = afrag<_Float16>(A, nullptr, K, mrow, k0 + 16 * grp);
#pragma unroll
        for (int t = 0; t < NT; ++t) {
            v16h b = bfrag(Wt, K, t * 16 + lid, k0 + 16 * grp);
            acc[t] = __builtin_amdgcn_wmma_f32_16x16x32_f16(false, a, false, b, (short)0,
                                                            acc[t], false, false);
        }
    }
#pragma unroll
    for (int r = 0; r < 8; ++r) {
        long m = strip * 16 + r + grp * 8;
        int  rj = rowi[m];
        int  ci = coli[m];
        float cc = ccos[m];
#pragma unroll
        for (int t = 0; t < NT; ++t) {
            int ncol = t * 16 + lid;
            float val = (acc[t][r] + bias[ncol]) * cc * hv[(long)rj * HIDD + ncol];
            atomicAdd(&agg[(long)ci * HIDD + ncol], val);
        }
    }
}

// GINE edge GEMM (K=32, 16x128 strip) with fused relu(h[csrc]+.) scatter-add to cdst.
__global__ void gemm_gine_scatter_kernel(const _Float16* __restrict__ Ea,
                                         const _Float16* __restrict__ Wt,
                                         const float* __restrict__ eb,
                                         const float* __restrict__ h,
                                         const int* __restrict__ csrc,
                                         const int* __restrict__ cdst,
                                         float* __restrict__ agg, long strips) {
    const int K = 32, NT = HIDD / 16;
    long strip = (long)blockIdx.x * 8 + (threadIdx.x >> 5);
    if (strip >= strips) return;
    int lane = threadIdx.x & 31;
    int grp = lane >> 4, lid = lane & 15;
    long mrow = strip * 16 + lid;

    const v8f zero = {};
    v8f acc[NT];
#pragma unroll
    for (int t = 0; t < NT; ++t) acc[t] = zero;

    v16h a = afrag<_Float16>(Ea, nullptr, K, mrow, 16 * grp);
#pragma unroll
    for (int t = 0; t < NT; ++t) {
        v16h b = bfrag(Wt, K, t * 16 + lid, 16 * grp);
        acc[t] = __builtin_amdgcn_wmma_f32_16x16x32_f16(false, a, false, b, (short)0,
                                                        acc[t], false, false);
    }
#pragma unroll
    for (int r = 0; r < 8; ++r) {
        long m = strip * 16 + r + grp * 8;
        int  sj = csrc[m];
        int  dj = cdst[m];
#pragma unroll
        for (int t = 0; t < NT; ++t) {
            int ncol = t * 16 + lid;
            float msg = fmaxf(h[(long)sj * HIDD + ncol] + acc[t][r] + eb[ncol], 0.f);
            atomicAdd(&agg[(long)dj * HIDD + ncol], msg);
        }
    }
}

// ---------------------------------------------------------------- elementwise / util

__global__ void edge_basis_kernel(const int* __restrict__ rowi, const int* __restrict__ coli,
                                  const float* __restrict__ pos, _Float16* __restrict__ demb,
                                  float* __restrict__ ccos) {
    int e = blockIdx.x * blockDim.x + threadIdx.x;
    if (e >= ESCH) return;
    int j = rowi[e], i = coli[e];
    float dx = pos[3 * j + 0] - pos[3 * i + 0];
    float dy = pos[3 * j + 1] - pos[3 * i + 1];
    float dz = pos[3 * j + 2] - pos[3 * i + 2];
    float d = sqrtf(dx * dx + dy * dy + dz * dz);
    ccos[e] = 0.5f * (__cosf(d * (PI_F / CUTOFF_R)) + 1.f);
    const float step = CUTOFF_R / (float)(NGAUSS - 1);
    const float coeff = -0.5f / (step * step);
    _Float16* out = demb + (long)e * 64;
#pragma unroll
    for (int g = 0; g < 64; ++g) {
        float val = 0.f;
        if (g < NGAUSS) { float t = d - (float)g * step; val = __expf(coeff * t * t); }
        out[g] = (_Float16)val;
    }
}

__global__ void init_v_kernel(const int* __restrict__ z, const float* __restrict__ tab,
                              float* __restrict__ v, long n) {
    long t = (long)blockIdx.x * blockDim.x + threadIdx.x;
    if (t >= n) return;
    long node = t >> 7;
    int c = (int)(t & 127);
    v[t] = tab[(long)z[node] * HIDD + c];
}

// W[K,N] f32 -> Wt[Npad,Kpad] f16 (transposed, zero padded)
__global__ void wconv_kernel(const float* __restrict__ W, _Float16* __restrict__ Wt,
                             int K, int N, int Kpad, int Npad) {
    long t = (long)blockIdx.x * blockDim.x + threadIdx.x;
    if (t >= (long)Npad * Kpad) return;
    int n = (int)(t / Kpad), k = (int)(t % Kpad);
    float val = (k < K && n < N) ? W[(long)k * N + n] : 0.f;
    Wt[t] = (_Float16)val;
}

__global__ void padb_kernel(const float* __restrict__ b, float* __restrict__ bp, int N, int Npad) {
    int t = blockIdx.x * blockDim.x + threadIdx.x;
    if (t < Npad) bp[t] = (t < N) ? b[t] : 0.f;
}

__global__ void pad_f32_to_f16_kernel(const float* __restrict__ src, _Float16* __restrict__ dst,
                                      long M, int K, int Kpad) {
    long t = (long)blockIdx.x * blockDim.x + threadIdx.x;
    if (t >= M * Kpad) return;
    long m = t / Kpad;
    int k = (int)(t % Kpad);
    dst[t] = (k < K) ? (_Float16)src[m * K + k] : (_Float16)0.f;
}

__global__ void add_kernel(const float* __restrict__ a, const float* __restrict__ b,
                           float* __restrict__ c, long n) {
    long t = (long)blockIdx.x * blockDim.x + threadIdx.x;
    if (t < n) c[t] = a[t] + b[t];
}

__global__ void segment_rows_kernel(const float* __restrict__ src, const int* __restrict__ idx,
                                    float* __restrict__ dst, long M, int N) {
    long t = (long)blockIdx.x * blockDim.x + threadIdx.x;
    if (t >= M * (long)N) return;
    long m = t / N;
    int n = (int)(t % N);
    atomicAdd(&dst[(long)idx[m] * N + n], src[t]);
}

__global__ void build_cat_kernel(const float* __restrict__ gp, const float* __restrict__ fc,
                                 const float* __restrict__ sp, _Float16* __restrict__ cat) {
    long t = (long)blockIdx.x * blockDim.x + threadIdx.x;
    if (t >= (long)NGRAPHS * 384) return;
    long g = t / 384;
    int c = (int)(t % 384);
    float val = (c < 128) ? gp[g * 128 + c]
              : (c < 256) ? fc[g * 128 + (c - 128)]
                          : sp[g * 128 + (c - 256)];
    cat[t] = (_Float16)val;
}

__global__ void copy_f32_kernel(const float* __restrict__ s, float* __restrict__ d, long n) {
    long t = (long)blockIdx.x * blockDim.x + threadIdx.x;
    if (t < n) d[t] = s[t];
}

__global__ void copy_out12_kernel(const float* __restrict__ o16, float* __restrict__ d) {
    long t = (long)blockIdx.x * blockDim.x + threadIdx.x;
    if (t >= (long)NGRAPHS * NTASKS) return;
    long g = t / NTASKS;
    int c = (int)(t % NTASKS);
    d[t] = o16[g * 16 + c];
}

// ---------------------------------------------------------------- host

static inline long divup_l(long a, long b) { return (a + b - 1) / b; }

#define GEMM(TA, TO, NT, A, A2, WT, BIAS, RES, D, M, K, ACT)                                       \
    do {                                                                                           \
        long strips_ = (long)(M) / 16;                                                             \
        gemm_kernel<TA, TO, NT><<<(unsigned)divup_l(strips_, 8), 256, 0, stream>>>(                \
            (A), (A2), (WT), (BIAS), (RES), (D), strips_, (K), (ACT));                             \
    } while (0)

extern "C" void kernel_launch(void* const* d_in, const int* in_sizes, int n_in,
                              void* d_out, int out_size, void* d_ws, size_t ws_size,
                              hipStream_t stream) {
    (void)in_sizes; (void)n_in; (void)out_size; (void)ws_size;

    const int*   z      = (const int*)  d_in[0];
    const float* pos    = (const float*)d_in[1];
    const int*   batch  = (const int*)  d_in[2];
    const int*   sch    = (const int*)  d_in[3];
    const float* xfeat  = (const float*)d_in[4];
    const int*   chem   = (const int*)  d_in[5];
    const float* eattr  = (const float*)d_in[6];
    const float* fp     = (const float*)d_in[7];
    const float* vtab   = (const float*)d_in[8];
    const float* ue_linv= (const float*)d_in[9];
    const float* ue_w1  = (const float*)d_in[10];
    const float* ue_b1  = (const float*)d_in[11];
    const float* ue_w2  = (const float*)d_in[12];
    const float* ue_b2  = (const float*)d_in[13];
    const float* uv_w1  = (const float*)d_in[14];
    const float* uv_b1  = (const float*)d_in[15];
    const float* uv_w2  = (const float*)d_in[16];
    const float* uv_b2  = (const float*)d_in[17];
    const float* uu_w1  = (const float*)d_in[18];
    const float* uu_b1  = (const float*)d_in[19];
    const float* uu_w2  = (const float*)d_in[20];
    const float* uu_b2  = (const float*)d_in[21];
    const float* n2h_w  = (const float*)d_in[22];
    const float* n2h_b  = (const float*)d_in[23];
    const float* nn_w   = (const float*)d_in[24];
    const float* nn_b   = (const float*)d_in[25];
    const float* g_ew   = (const float*)d_in[26];
    const float* g_eb   = (const float*)d_in[27];
    const float* fpn_w1 = (const float*)d_in[28];
    const float* fpn_b1 = (const float*)d_in[29];
    const float* fpn_w2 = (const float*)d_in[30];
    const float* fpn_b2 = (const float*)d_in[31];
    const float* hid_w  = (const float*)d_in[32];
    const float* hid_b  = (const float*)d_in[33];
    const float* out_w  = (const float*)d_in[34];
    const float* out_b  = (const float*)d_in[35];

    const int* rowi = sch;             // source j
    const int* coli = sch + ESCH;      // target i
    const int* csrc = chem;
    const int* cdst = chem + ECHEM;

    // ---- workspace bump allocator (256B aligned)
    char* base = (char*)d_ws;
    size_t off = 0;
    auto alloc = [&](size_t bytes) -> void* {
        off = (off + 255) & ~(size_t)255;
        void* p = base + off;
        off += bytes;
        return p;
    };

    _Float16* demb  = (_Float16*)alloc((size_t)ESCH * 64 * 2);
    float*    ccos  = (float*)   alloc((size_t)ESCH * 4);
    _Float16* wtmp  = (_Float16*)alloc((size_t)ESCH * HIDD * 2);
    float*    v     = (float*)   alloc((size_t)NNODES * HIDD * 4);
    float*    hv    = (float*)   alloc((size_t)NNODES * HIDD * 4);
    float*    agg   = (float*)   alloc((size_t)NNODES * HIDD * 4);
    _Float16* t1h   = (_Float16*)alloc((size_t)NNODES * HIDD * 2);
    float*    hidA  = (float*)   alloc((size_t)NNODES * HIDD * 4);
    float*    hidB  = (float*)   alloc((size_t)NNODES * HIDD * 4);
    float*    agg2  = (float*)   alloc((size_t)NNODES * HIDD * 4);
    float*    sn    = (float*)   alloc((size_t)NNODES * HIDD * 4);
    float*    vf    = (float*)   alloc((size_t)NNODES * HIDD * 4);
    _Float16* uh1h  = (_Float16*)alloc((size_t)NNODES * 64 * 2);
    float*    uh2   = (float*)   alloc((size_t)NNODES * HIDD * 4);
    float*    gl    = (float*)   alloc((size_t)NNODES * HIDD * 4);
    _Float16* xfh   = (_Float16*)alloc((size_t)NNODES * 64 * 2);
    _Float16* eah   = (_Float16*)alloc((size_t)ECHEM * 32 * 2);
    _Float16* f1h   = (_Float16*)alloc((size_t)NGRAPHS * HIDD * 2);
    float*    fcn   = (float*)   alloc((size_t)NGRAPHS * HIDD * 4);
    float*    gp    = (float*)   alloc((size_t)NGRAPHS * HIDD * 4);
    float*    sp    = (float*)   alloc((size_t)NGRAPHS * HIDD * 4);
    _Float16* cath  = (_Float16*)alloc((size_t)NGRAPHS * 384 * 2);
    _Float16* h1h   = (_Float16*)alloc((size_t)NGRAPHS * HIDD * 2);
    float*    o16   = (float*)   alloc((size_t)NGRAPHS * 16 * 4);

    _Float16* linvT = (_Float16*)alloc((size_t)6 * 128 * 128 * 2);
    _Float16* w1T   = (_Float16*)alloc((size_t)6 * 128 * 64 * 2);
    _Float16* w2T   = (_Float16*)alloc((size_t)6 * 128 * 128 * 2);
    _Float16* uvw1T = (_Float16*)alloc((size_t)6 * 128 * 128 * 2);
    _Float16* uvw2T = (_Float16*)alloc((size_t)6 * 128 * 128 * 2);
    _Float16* ewT   = (_Float16*)alloc((size_t)6 * 128 * 32 * 2);
    _Float16* uuw1T = (_Float16*)alloc((size_t)64 * 128 * 2);
    _Float16* uuw2T = (_Float16*)alloc((size_t)128 * 64 * 2);
    _Float16* n2hT  = (_Float16*)alloc((size_t)128 * 64 * 2);
    _Float16* nnT   = (_Float16*)alloc((size_t)128 * 128 * 2);
    _Float16* fpn1T = (_Float16*)alloc((size_t)128 * 2048 * 2);
    _Float16* fpn2T = (_Float16*)alloc((size_t)128 * 128 * 2);
    _Float16* hidT  = (_Float16*)alloc((size_t)128 * 384 * 2);
    _Float16* outT  = (_Float16*)alloc((size_t)16 * 128 * 2);
    float*    outbp = (float*)   alloc((size_t)16 * 4);

    auto wc = [&](const float* W, _Float16* Wt, int K, int N, int Kpad, int Npad) {
        long tot = (long)Npad * Kpad;
        wconv_kernel<<<(unsigned)divup_l(tot, 256), 256, 0, stream>>>(W, Wt, K, N, Kpad, Npad);
    };

    // ---- weight prep (f16, transposed, padded)
    for (int i = 0; i < 6; ++i) {
        wc(ue_linv + (long)i * 128 * 128, linvT + (long)i * 128 * 128, 128, 128, 128, 128);
        wc(ue_w1   + (long)i * 50 * 128,  w1T   + (long)i * 128 * 64,  50,  128, 64,  128);
        wc(ue_w2   + (long)i * 128 * 128, w2T   + (long)i * 128 * 128, 128, 128, 128, 128);
        wc(uv_w1   + (long)i * 128 * 128, uvw1T + (long)i * 128 * 128, 128, 128, 128, 128);
        wc(uv_w2   + (long)i * 128 * 128, uvw2T + (long)i * 128 * 128, 128, 128, 128, 128);
        wc(g_ew    + (long)i * 10 * 128,  ewT   + (long)i * 128 * 32,  10,  128, 32,  128);
    }
    wc(uu_w1, uuw1T, 128, 64, 128, 64);
    wc(uu_w2, uuw2T, 64, 128, 64, 128);
    wc(n2h_w, n2hT, NODEDIM, 128, 64, 128);
    wc(nn_w, nnT, 128, 128, 128, 128);
    wc(fpn_w1, fpn1T, 2048, 128, 2048, 128);
    wc(fpn_w2, fpn2T, 128, 128, 128, 128);
    wc(hid_w, hidT, 384, 128, 384, 128);
    wc(out_w, outT, 128, NTASKS, 128, 16);
    padb_kernel<<<1, 32, 0, stream>>>(out_b, outbp, NTASKS, 16);

    // ---- input prep
    pad_f32_to_f16_kernel<<<(unsigned)divup_l((long)NNODES * 64, 256), 256, 0, stream>>>(
        xfeat, xfh, NNODES, NODEDIM, 64);
    pad_f32_to_f16_kernel<<<(unsigned)divup_l((long)ECHEM * 32, 256), 256, 0, stream>>>(
        eattr, eah, ECHEM, BONDDIM, 32);
    edge_basis_kernel<<<(unsigned)divup_l(ESCH, 256), 256, 0, stream>>>(rowi, coli, pos, demb, ccos);
    init_v_kernel<<<(unsigned)divup_l((long)NNODES * HIDD, 256), 256, 0, stream>>>(
        z, vtab, v, (long)NNODES * HIDD);

    // hid0 = x_feat @ node2hid + b
    GEMM(_Float16, float, 8, xfh, (const _Float16*)nullptr, n2hT, n2h_b, (const float*)nullptr,
         hidA, NNODES, 64, 0);

    const size_t nodeBytes = (size_t)NNODES * HIDD * 4;
    long estrips = (long)ESCH / 16;
    long cstrips = (long)ECHEM / 16;

    float* hcur = hidA;
    float* hnxt = hidB;

    for (int i = 0; i < NLAYERS - 1; ++i) {
        // -------- SchNet interaction i
        GEMM(float, float, 8, v, (const float*)nullptr, linvT + (long)i * 128 * 128,
             (const float*)nullptr, (const float*)nullptr, hv, NNODES, HIDD, 0);
        hipMemsetAsync(agg, 0, nodeBytes, stream);
        GEMM(_Float16, _Float16, 8, demb, (const _Float16*)nullptr, w1T + (long)i * 128 * 64,
             ue_b1 + (long)i * 128, (const float*)nullptr, wtmp, ESCH, 64, 2);
        gemm_filter_scatter_kernel<<<(unsigned)divup_l(estrips, 8), 256, 0, stream>>>(
            wtmp, w2T + (long)i * 128 * 128, ue_b2 + (long)i * 128, ccos, hv, rowi, coli, agg,
            estrips);
        GEMM(float, _Float16, 8, agg, (const float*)nullptr, uvw1T + (long)i * 128 * 128,
             uv_b1 + (long)i * 128, (const float*)nullptr, t1h, NNODES, HIDD, 2);
        GEMM(_Float16, float, 8, t1h, (const _Float16*)nullptr, uvw2T + (long)i * 128 * 128,
             uv_b2 + (long)i * 128, v, v, NNODES, HIDD, 0);

        // -------- GINE layer i
        hipMemsetAsync(agg2, 0, nodeBytes, stream);
        gemm_gine_scatter_kernel<<<(unsigned)divup_l(cstrips, 8), 256, 0, stream>>>(
            eah, ewT + (long)i * 128 * 32, g_eb + (long)i * 128, hcur, csrc, cdst, agg2, cstrips);
        GEMM(float, float, 8, hcur, agg2, nnT, nn_b, (const float*)nullptr, hnxt, NNODES,
             HIDD, 1);
        float* t = hcur; hcur = hnxt; hnxt = t;
    }

    // -------- fusion: sch_nodes = v + hid
    add_kernel<<<(unsigned)divup_l((long)NNODES * HIDD, 256), 256, 0, stream>>>(
        v, hcur, sn, (long)NNODES * HIDD);

    // -------- final SchNet layer (i = 5) on sch_nodes
    {
        int i = NLAYERS - 1;
        GEMM(float, float, 8, sn, (const float*)nullptr, linvT + (long)i * 128 * 128,
             (const float*)nullptr, (const float*)nullptr, hv, NNODES, HIDD, 0);
        hipMemsetAsync(agg, 0, nodeBytes, stream);
        GEMM(_Float16, _Float16, 8, demb, (const _Float16*)nullptr, w1T + (long)i * 128 * 64,
             ue_b1 + (long)i * 128, (const float*)nullptr, wtmp, ESCH, 64, 2);
        gemm_filter_scatter_kernel<<<(unsigned)divup_l(estrips, 8), 256, 0, stream>>>(
            wtmp, w2T + (long)i * 128 * 128, ue_b2 + (long)i * 128, ccos, hv, rowi, coli, agg,
            estrips);
        GEMM(float, _Float16, 8, agg, (const float*)nullptr, uvw1T + (long)i * 128 * 128,
             uv_b1 + (long)i * 128, (const float*)nullptr, t1h, NNODES, HIDD, 2);
        GEMM(_Float16, float, 8, t1h, (const _Float16*)nullptr, uvw2T + (long)i * 128 * 128,
             uv_b2 + (long)i * 128, sn, vf, NNODES, HIDD, 0);
    }

    // update_u -> sch_pool
    GEMM(float, _Float16, 4, vf, (const float*)nullptr, uuw1T, uu_b1, (const float*)nullptr, uh1h,
         NNODES, HIDD, 2);
    GEMM(_Float16, float, 8, uh1h, (const _Float16*)nullptr, uuw2T, uu_b2, (const float*)nullptr,
         uh2, NNODES, 64, 0);
    hipMemsetAsync(sp, 0, (size_t)NGRAPHS * HIDD * 4, stream);
    segment_rows_kernel<<<(unsigned)divup_l((long)NNODES * HIDD, 256), 256, 0, stream>>>(
        uh2, batch, sp, NNODES, HIDD);

    // final GINE on sch_nodes -> gnn_pool
    hipMemsetAsync(agg2, 0, nodeBytes, stream);
    {
        int i = NLAYERS - 1;
        gemm_gine_scatter_kernel<<<(unsigned)divup_l(cstrips, 8), 256, 0, stream>>>(
            eah, ewT + (long)i * 128 * 32, g_eb + (long)i * 128, sn, csrc, cdst, agg2, cstrips);
    }
    GEMM(float, float, 8, sn, agg2, nnT, nn_b, (const float*)nullptr, gl, NNODES, HIDD, 1);
    hipMemsetAsync(gp, 0, (size_t)NGRAPHS * HIDD * 4, stream);
    segment_rows_kernel<<<(unsigned)divup_l((long)NNODES * HIDD, 256), 256, 0, stream>>>(
        gl, batch, gp, NNODES, HIDD);

    // FPN
    GEMM(float, _Float16, 8, fp, (const float*)nullptr, fpn1T, fpn_b1, (const float*)nullptr, f1h,
         NGRAPHS, FPSIZE, 1);
    GEMM(_Float16, float, 8, f1h, (const _Float16*)nullptr, fpn2T, fpn_b2, (const float*)nullptr,
         fcn, NGRAPHS, HIDD, 0);

    // head: concat -> relu(hid) -> out
    build_cat_kernel<<<(unsigned)divup_l((long)NGRAPHS * 384, 256), 256, 0, stream>>>(
        gp, fcn, sp, cath);
    GEMM(_Float16, _Float16, 8, cath, (const _Float16*)nullptr, hidT, hid_b, (const float*)nullptr,
         h1h, NGRAPHS, 384, 1);
    GEMM(_Float16, float, 1, h1h, (const _Float16*)nullptr, outT, outbp, (const float*)nullptr,
         o16, NGRAPHS, HIDD, 0);

    // outputs: (out[2000,12], gnn_pool[2000,128], sch_pool[2000,128]) flat
    float* dout = (float*)d_out;
    copy_out12_kernel<<<(unsigned)divup_l((long)NGRAPHS * NTASKS, 256), 256, 0, stream>>>(o16, dout);
    copy_f32_kernel<<<(unsigned)divup_l((long)NGRAPHS * HIDD, 256), 256, 0, stream>>>(
        gp, dout + (long)NGRAPHS * NTASKS, (long)NGRAPHS * HIDD);
    copy_f32_kernel<<<(unsigned)divup_l((long)NGRAPHS * HIDD, 256), 256, 0, stream>>>(
        sp, dout + (long)NGRAPHS * (NTASKS + HIDD), (long)NGRAPHS * HIDD);
}
